// SCNN_20134806683713
// MI455X (gfx1250) — compile-verified, matrix operand order checked by reading, not run on hardware
//
#include <hip/hip_runtime.h>
#include <hip/hip_bf16.h>

typedef __attribute__((ext_vector_type(16))) __bf16 v16bf;
typedef __attribute__((ext_vector_type(8)))  __bf16 v8bf;
typedef __attribute__((ext_vector_type(8)))  float  v8f;

#define BB   16
#define CC   128
#define HH   64
#define WW   160
#define KWIN 9
#define PADK 4
#define KTOT (KWIN * CC)   // 1152
#define KB   (KTOT / 32)   // 36 K-blocks of 32
#define MT   32            // M columns per block
#define HALO (2 * PADK)    // 8
#define SROWS (MT + HALO)  // 40 staged rows

union Frag16 { v16bf v; struct { v8bf lo, hi; } h; };

// ---------------- layout transform kernels ----------------

// x (B,C,H,W) f32  ->  Y (B,H,W,C) f32
__global__ void pack_bhwc(const float* __restrict__ x, float* __restrict__ y) {
    size_t i = (size_t)blockIdx.x * 256 + threadIdx.x;
    const size_t total = (size_t)BB * CC * HH * WW;
    if (i >= total) return;
    int c = (int)(i % CC);  size_t t = i / CC;
    int w = (int)(t % WW);  t /= WW;
    int h = (int)(t % HH);  int b = (int)(t / HH);
    y[i] = x[(((size_t)b * CC + c) * HH + h) * WW + w];
}

// Y (B,H,W,C) f32 -> out (B,C,H,W) f32
__global__ void unpack_bchw(const float* __restrict__ y, float* __restrict__ out) {
    size_t i = (size_t)blockIdx.x * 256 + threadIdx.x;
    const size_t total = (size_t)BB * CC * HH * WW;
    if (i >= total) return;
    int w = (int)(i % WW);  size_t t = i / WW;
    int h = (int)(t % HH);  t /= HH;
    int c = (int)(t % CC);  int b = (int)(t / CC);
    out[i] = y[(((size_t)b * HH + h) * WW + w) * CC + c];
}

// weights f32 (C_out, 1, K, C_in) -> bf16 [C_out][K*C_in] (same flat order)
__global__ void cvt_weights(const float* __restrict__ src, __bf16* __restrict__ dst, int n) {
    int i = blockIdx.x * 256 + threadIdx.x;
    if (i < n) dst[i] = (__bf16)src[i];
}

// ---------------- one sweep step: Y[p] += relu(conv(Y[pprev])) ----------------
// Generalized over axes: Y[b][p][q][c] with strides strideP/strideQ (in floats).
// Per-block GEMM tile: M = 32 q-positions, N = 128 channels, K = 9*128.
__global__ __launch_bounds__(256)
void sweep_step(float* __restrict__ Y, const __bf16* __restrict__ Wt,
                int p, int pprev, int strideP, int strideQ, int Q) {
    __shared__ __align__(16) __bf16 smem[SROWS * CC];   // 10240 B

    const int b  = blockIdx.y;
    const int q0 = blockIdx.x * MT;
    const size_t bOff = (size_t)b * HH * WW * CC;
    const float* prevBase = Y + bOff + (size_t)pprev * strideP;

    // Stage prev line (with halo) into LDS as bf16
    for (int idx = threadIdx.x; idx < SROWS * CC; idx += 256) {
        int i = idx >> 7;        // staged row 0..39
        int c = idx & 127;
        int q = q0 + i - PADK;
        float v = (q >= 0 && q < Q) ? prevBase[(size_t)q * strideQ + c] : 0.0f;
        smem[idx] = (__bf16)v;
    }
    __syncthreads();

    const int wave  = threadIdx.x >> 5;
    const int lane  = threadIdx.x & 31;
    const int mBase = (wave >> 2) * 16;      // 0 or 16
    const int nBase = (wave & 3) * 32;       // 0,32,64,96
    const int lm    = lane & 15;
    const int kh    = (lane >> 4) * 8;       // K sub-offset per lane half

    v8f acc0 = {}; v8f acc1 = {};
    const __bf16* w0 = Wt + (size_t)(nBase + lm) * KTOT;
    const __bf16* w1 = Wt + (size_t)(nBase + 16 + lm) * KTOT;

    for (int kb = 0; kb < KB; ++kb) {
        const int tap = kb >> 2;
        const int ciB = (kb & 3) << 5;

        // A fragment: 16x32 bf16, lane lm = matrix row m, 8+8 consecutive K
        const __bf16* arow = smem + (mBase + lm + tap) * CC + ciB + kh;
        Frag16 A;
        A.h.lo = *(const v8bf*)(arow);
        A.h.hi = *(const v8bf*)(arow + 16);

        // B fragments: 32x16 bf16, lane lm = matrix col n, same K striping
        const int ko = kb * 32 + kh;
        Frag16 B0, B1;
        B0.h.lo = *(const v8bf*)(w0 + ko);
        B0.h.hi = *(const v8bf*)(w0 + ko + 16);
        B1.h.lo = *(const v8bf*)(w1 + ko);
        B1.h.hi = *(const v8bf*)(w1 + ko + 16);

        acc0 = __builtin_amdgcn_wmma_f32_16x16x32_bf16(false, A.v, false, B0.v,
                                                       (short)0, acc0, false, false);
        acc1 = __builtin_amdgcn_wmma_f32_16x16x32_bf16(false, A.v, false, B1.v,
                                                       (short)0, acc1, false, false);
    }

    // Epilogue: Y[p] += relu(acc). C/D layout: VGPR v -> M=v (lanes 0-15) / v+8.
    float* cur = Y + bOff + (size_t)p * strideP;
    const int mLane = mBase + ((lane < 16) ? 0 : 8);
#pragma unroll
    for (int v = 0; v < 8; ++v) {
        const int q = q0 + mLane + v;
        float* o0 = cur + (size_t)q * strideQ + nBase + lm;
        float r0 = acc0[v]; r0 = r0 > 0.0f ? r0 : 0.0f;
        o0[0] += r0;
        float r1 = acc1[v]; r1 = r1 > 0.0f ? r1 : 0.0f;
        o0[16] += r1;
    }
}

// ---------------- host side ----------------

extern "C" void kernel_launch(void* const* d_in, const int* in_sizes, int n_in,
                              void* d_out, int out_size, void* d_ws, size_t ws_size,
                              hipStream_t stream) {
    const float* x   = (const float*)d_in[0];
    const float* ktd = (const float*)d_in[1];
    const float* kdt = (const float*)d_in[2];
    const float* klr = (const float*)d_in[3];
    const float* krl = (const float*)d_in[4];

    const size_t yElems = (size_t)BB * HH * WW * CC;
    float*  Y   = (float*)d_ws;
    __bf16* wts = (__bf16*)((char*)d_ws + yElems * sizeof(float));
    const int WN = CC * KWIN * CC;   // 147456 per direction

    dim3 blk(256);
    const int gW = (WN + 255) / 256;
    cvt_weights<<<gW, blk, 0, stream>>>(ktd, wts + (size_t)0 * WN, WN);
    cvt_weights<<<gW, blk, 0, stream>>>(kdt, wts + (size_t)1 * WN, WN);
    cvt_weights<<<gW, blk, 0, stream>>>(klr, wts + (size_t)2 * WN, WN);
    cvt_weights<<<gW, blk, 0, stream>>>(krl, wts + (size_t)3 * WN, WN);

    const int gY = (int)((yElems + 255) / 256);
    pack_bhwc<<<gY, blk, 0, stream>>>(x, Y);

    // Row sweeps: p axis = H (stride W*C), q/M axis = W (stride C), conv over W
    dim3 gridRow(WW / MT, BB);   // (5,16)
    for (int h = 1; h < HH; ++h)
        sweep_step<<<gridRow, blk, 0, stream>>>(Y, wts + (size_t)0 * WN,
                                                h, h - 1, WW * CC, CC, WW);
    for (int h = HH - 2; h >= 0; --h)
        sweep_step<<<gridRow, blk, 0, stream>>>(Y, wts + (size_t)1 * WN,
                                                h, h + 1, WW * CC, CC, WW);

    // Column sweeps: p axis = W (stride C), q/M axis = H (stride W*C), conv over H
    dim3 gridCol(HH / MT, BB);   // (2,16)
    for (int w = 1; w < WW; ++w)
        sweep_step<<<gridCol, blk, 0, stream>>>(Y, wts + (size_t)2 * WN,
                                                w, w - 1, CC, WW * CC, HH);
    for (int w = WW - 2; w >= 0; --w)
        sweep_step<<<gridCol, blk, 0, stream>>>(Y, wts + (size_t)3 * WN,
                                                w, w + 1, CC, WW * CC, HH);

    unpack_bchw<<<gY, blk, 0, stream>>>(Y, (float*)d_out);
}
